// HLLMTransformerBlock_9594956939299
// MI455X (gfx1250) — matrix operand order repaired and checked by reading, not compiled, not run on hardware
//
#include <hip/hip_runtime.h>
#include <hip/hip_bf16.h>
#include <stdint.h>

#define D_MODEL  1024
#define N_HEADS  16
#define HEAD_DIM 64
#define FFN_H    4096
#define BATCH    4
#define SEQL     2048
#define NROWS    (BATCH * SEQL)   // 8192

typedef __attribute__((ext_vector_type(16))) __bf16 v16bf;
typedef __attribute__((ext_vector_type(8)))  float  v8f;

union FragU { uint4 u[2]; v16bf v; };

__device__ __forceinline__ unsigned short f2bf(float f) {
  unsigned int u = __builtin_bit_cast(unsigned int, f);
  return (unsigned short)((u + 0x7fffu + ((u >> 16) & 1u)) >> 16);  // RNE
}

// A fragment: 16x32 bf16 (MxK).  lane<16: elems[0..7]=K k0..k0+7, [8..15]=k0+16..23
//             lane>=16: shifted by 8 in K.  Two 16B contiguous chunks per lane.
__device__ __forceinline__ v16bf load_frag_a(const unsigned short* base, int ld,
                                             int m0, int k0, int lane) {
  const int hi = (lane >> 4) & 1;
  const int m  = lane & 15;
  const unsigned short* p = base + (size_t)(m0 + m) * ld + k0 + hi * 8;
  FragU f;
  f.u[0] = *reinterpret_cast<const uint4*>(p);
  f.u[1] = *reinterpret_cast<const uint4*>(p + 16);
  return f.v;
}

// B fragment: 32x16 bf16 (KxN) taken from a row-major [N,K] matrix (i.e. B = W^T).
// lane&15 = column n; lanes 0-15: elems[0..7]=k0..k0+7, [8..15]=k0+16..23; lanes 16-31 shifted by 8.
__device__ __forceinline__ v16bf load_frag_b(const unsigned short* base, int ld,
                                             int n0, int k0, int lane) {
  const int hi = (lane >> 4) & 1;
  const int n  = lane & 15;
  const unsigned short* p = base + (size_t)(n0 + n) * ld + k0 + hi * 8;
  FragU f;
  f.u[0] = *reinterpret_cast<const uint4*>(p);
  f.u[1] = *reinterpret_cast<const uint4*>(p + 16);
  return f.v;
}

__device__ __forceinline__ v8f wmma_bf16(v16bf a, v16bf b, v8f c) {
  return __builtin_amdgcn_wmma_f32_16x16x32_bf16(false, a, false, b, (short)0, c,
                                                 false, false);
}

// ---------------------------------------------------------------- converts
__global__ void f32_to_bf16_kernel(const float* __restrict__ in,
                                   unsigned short* __restrict__ out, int n) {
  int i = blockIdx.x * blockDim.x + threadIdx.x;
  if (i < n) out[i] = f2bf(in[i]);
}

// ---------------------------------------------------------------- rmsnorm
__global__ void rmsnorm_kernel(const float* __restrict__ x,
                               const float* __restrict__ w,
                               unsigned short* __restrict__ out) {
  __shared__ float red[8];
  __shared__ float inv_s;
  const int row = blockIdx.x;
  const int tid = threadIdx.x;           // 256 threads, 4 floats each
  const float* xr = x + (size_t)row * D_MODEL;
  float4 v = reinterpret_cast<const float4*>(xr)[tid];
  float ss = v.x * v.x + v.y * v.y + v.z * v.z + v.w * v.w;
#pragma unroll
  for (int off = 16; off > 0; off >>= 1) ss += __shfl_down(ss, off, 32);
  if ((tid & 31) == 0) red[tid >> 5] = ss;
  __syncthreads();
  if (tid == 0) {
    float t = 0.f;
#pragma unroll
    for (int i = 0; i < 8; ++i) t += red[i];
    inv_s = rsqrtf(t * (1.0f / D_MODEL) + 1e-6f);
  }
  __syncthreads();
  const float inv = inv_s;
  float4 wv = reinterpret_cast<const float4*>(w)[tid];
  unsigned short o[4];
  o[0] = f2bf(v.x * inv * wv.x);
  o[1] = f2bf(v.y * inv * wv.y);
  o[2] = f2bf(v.z * inv * wv.z);
  o[3] = f2bf(v.w * inv * wv.w);
  *reinterpret_cast<uint2*>(out + (size_t)row * D_MODEL + tid * 4) =
      *reinterpret_cast<uint2*>(o);
}

// ---------------------------------------------------------------- GEMM
// Block-tiled: 128(M) x 64(N) per block, 8 waves in 4x2 grid, each wave a
// 32x32 register tile (4 accumulators, 4 wmma per 32-wide K-step).
// K slabs staged through double-buffered LDS; next slab prefetched into
// registers while the current one is consumed (1 barrier per K-step).
// out[m,n] = act( A[m,:] . W[n,:] + bias[n] ) (+ resid[m,n])
template <int OUT_BF16, int SILU, int RESID>
__global__ __launch_bounds__(256)
void gemm_bf16_tiled(const unsigned short* __restrict__ A,
                     const unsigned short* __restrict__ W,
                     const float* __restrict__ bias,
                     const float* __restrict__ resid,
                     void* __restrict__ out, int M, int N, int K) {
  __shared__ unsigned short As[2][128][40];  // row stride 80B (16B aligned)
  __shared__ unsigned short Ws[2][64][40];

  const int tid  = threadIdx.x;
  const int lane = tid & 31;
  const int wid  = tid >> 5;
  const int wr   = wid >> 1;   // wave row 0..3  -> M offset 32*wr
  const int wc   = wid & 1;    // wave col 0..1  -> N offset 32*wc

  const int nblkn = N >> 6;
  const int m0 = ((int)blockIdx.x / nblkn) * 128;
  const int n0 = ((int)blockIdx.x % nblkn) * 64;

  // global staging: A 128x32 by 256 thr (16 elems each); W 64x32 by thr<128
  const int arow = tid >> 1, aseg = (tid & 1) * 16;
  const int wrow = (tid & 127) >> 1, wseg = (tid & 1) * 16;
  const unsigned short* agp = A + (size_t)(m0 + arow) * K + aseg;
  const unsigned short* wgp = W + (size_t)(n0 + wrow) * K + wseg;

  uint4 ra0 = {}, ra1 = {}, rw0 = {}, rw1 = {};
  ra0 = *reinterpret_cast<const uint4*>(agp);
  ra1 = *reinterpret_cast<const uint4*>(agp + 8);
  if (tid < 128) {
    rw0 = *reinterpret_cast<const uint4*>(wgp);
    rw1 = *reinterpret_cast<const uint4*>(wgp + 8);
  }

  v8f acc00 = {}, acc01 = {}, acc10 = {}, acc11 = {};
  int cur = 0;
  for (int k0 = 0; k0 < K; k0 += 32) {
    // commit staged slab to LDS[cur]
    *reinterpret_cast<uint4*>(&As[cur][arow][aseg])     = ra0;
    *reinterpret_cast<uint4*>(&As[cur][arow][aseg + 8]) = ra1;
    if (tid < 128) {
      *reinterpret_cast<uint4*>(&Ws[cur][wrow][wseg])     = rw0;
      *reinterpret_cast<uint4*>(&Ws[cur][wrow][wseg + 8]) = rw1;
    }
    __syncthreads();
    // prefetch next slab into registers (overlaps with wmma below)
    if (k0 + 32 < K) {
      ra0 = *reinterpret_cast<const uint4*>(agp + k0 + 32);
      ra1 = *reinterpret_cast<const uint4*>(agp + k0 + 40);
      if (tid < 128) {
        rw0 = *reinterpret_cast<const uint4*>(wgp + k0 + 32);
        rw1 = *reinterpret_cast<const uint4*>(wgp + k0 + 40);
      }
      __builtin_prefetch(agp + k0 + 64, 0, 0);  // speculative, next-next slab
    }
    // consume slab: 4 fragments -> 4 wmma (each frag feeds 2 wmma)
    v16bf a0 = load_frag_a(&As[cur][0][0], 40, wr * 32,      0, lane);
    v16bf a1 = load_frag_a(&As[cur][0][0], 40, wr * 32 + 16, 0, lane);
    v16bf b0 = load_frag_b(&Ws[cur][0][0], 40, wc * 32,      0, lane);
    v16bf b1 = load_frag_b(&Ws[cur][0][0], 40, wc * 32 + 16, 0, lane);
    acc00 = wmma_bf16(a0, b0, acc00);
    acc01 = wmma_bf16(a0, b1, acc01);
    acc10 = wmma_bf16(a1, b0, acc10);
    acc11 = wmma_bf16(a1, b1, acc11);
    cur ^= 1;
  }

  // epilogue
  const int hi = lane >> 4, nn = lane & 15;
  auto emit = [&](float v, int row, int col) {
    v += bias[col];
    if (SILU) v = v / (1.0f + __expf(-v));  // x * sigmoid(x)
    if (RESID) v += resid[(size_t)row * N + col];
    if (OUT_BF16)
      reinterpret_cast<unsigned short*>(out)[(size_t)row * N + col] = f2bf(v);
    else
      reinterpret_cast<float*>(out)[(size_t)row * N + col] = v;
  };
  const int r0 = m0 + wr * 32 + hi * 8;
  const int c0 = n0 + wc * 32 + nn;
#pragma unroll
  for (int r = 0; r < 8; ++r) {
    emit(acc00[r], r0 + r,      c0);
    emit(acc01[r], r0 + r,      c0 + 16);
    emit(acc10[r], r0 + r + 16, c0);
    emit(acc11[r], r0 + r + 16, c0 + 16);
  }
}

// ---------------------------------------------------------------- attention
// Flash-attention: 64 query rows per block (256 thr = 8 waves; 4x2 wave grid),
// online softmax in LDS, S=Q.K^T and O+=P.V on WMMA.
__global__ __launch_bounds__(256)
void attn_kernel(const unsigned short* __restrict__ Q,
                 const unsigned short* __restrict__ Kb,
                 const unsigned short* __restrict__ Vb,
                 const float* __restrict__ rpb,   // [H,L,L]
                 unsigned short* __restrict__ O) {
  __shared__ float          Sf[64][65];
  __shared__ unsigned short P[64][72];   // row stride 144B (16B aligned)
  __shared__ unsigned short Vt[64][72];  // V transposed: [hd][key]
  __shared__ float mrow[64], lrow[64], crow[64];

  const int tid  = threadIdx.x;
  const int lane = tid & 31;
  const int wid  = tid >> 5;
  const int wm   = wid >> 1;   // 0..3 -> query rows 16*wm
  const int wn   = wid & 1;    // 0..1 -> 32-col half
  const int hi   = lane >> 4;
  const int nn   = lane & 15;

  const int qb = blockIdx.x, h = blockIdx.y, b = blockIdx.z;
  const int q0 = qb * 64;
  const size_t rowbase = (size_t)b * SEQL;
  const unsigned short* qbase = Q + (rowbase + q0) * D_MODEL + h * HEAD_DIM;
  const float* bias_h = rpb + ((size_t)h * SEQL + q0) * SEQL;
  const float scale = 0.125f;  // hd^-0.5

  // Q fragments held in registers for the whole key loop
  v16bf qa0 = load_frag_a(qbase, D_MODEL, wm * 16, 0, lane);
  v16bf qa1 = load_frag_a(qbase, D_MODEL, wm * 16, 32, lane);

  v8f o0 = {}, o1 = {};
  if (tid < 64) { mrow[tid] = -3.0e38f; lrow[tid] = 0.f; }

  for (int kb = 0; kb <= qb; ++kb) {
    __syncthreads();  // protect Vt/P/Sf reuse across iterations

    // ---- stage V^T into LDS: 64 keys x 64 hd
    {
      const int krow = tid >> 2;   // key 0..63
      const int seg  = tid & 3;    // 16-elem hd segment
      const unsigned short* vp =
          Vb + (rowbase + kb * 64 + krow) * D_MODEL + h * HEAD_DIM + seg * 16;
      unsigned short tmp[16];
      *reinterpret_cast<uint4*>(tmp)     = *reinterpret_cast<const uint4*>(vp);
      *reinterpret_cast<uint4*>(tmp + 8) = *reinterpret_cast<const uint4*>(vp + 8);
#pragma unroll
      for (int e = 0; e < 16; ++e) Vt[seg * 16 + e][krow] = tmp[e];
    }

    // ---- S = Q . K^T  (K frags straight from global)
    const unsigned short* kbase =
        Kb + (rowbase + kb * 64) * D_MODEL + h * HEAD_DIM;
    v8f s0 = {}, s1 = {};
    {
      v16bf fb = load_frag_b(kbase, D_MODEL, wn * 32, 0, lane);
      s0 = wmma_bf16(qa0, fb, s0);
      fb = load_frag_b(kbase, D_MODEL, wn * 32, 32, lane);
      s0 = wmma_bf16(qa1, fb, s0);
      fb = load_frag_b(kbase, D_MODEL, wn * 32 + 16, 0, lane);
      s1 = wmma_bf16(qa0, fb, s1);
      fb = load_frag_b(kbase, D_MODEL, wn * 32 + 16, 32, lane);
      s1 = wmma_bf16(qa1, fb, s1);
    }

    // ---- scale + rel-pos bias + causal mask -> Sf
#pragma unroll
    for (int r = 0; r < 8; ++r) {
      const int row = wm * 16 + r + hi * 8;
      const int c0 = wn * 32 + nn;
      const int c1 = c0 + 16;
      float v0 = s0[r] * scale + bias_h[(size_t)row * SEQL + kb * 64 + c0];
      float v1 = s1[r] * scale + bias_h[(size_t)row * SEQL + kb * 64 + c1];
      if (kb == qb) {
        if (kb * 64 + c0 > q0 + row) v0 = -3.0e38f;
        if (kb * 64 + c1 > q0 + row) v1 = -3.0e38f;
      }
      Sf[row][c0] = v0;
      Sf[row][c1] = v1;
    }
    __syncthreads();

    // ---- online softmax per row
    if (tid < 64) {
      const float mo = mrow[tid];
      float mx = mo;
#pragma unroll 8
      for (int c = 0; c < 64; ++c) mx = fmaxf(mx, Sf[tid][c]);
      const float corr = (mo <= -2.9e38f) ? 0.f : __expf(mo - mx);
      float lsum = 0.f;
#pragma unroll 8
      for (int c = 0; c < 64; ++c) {
        const float p = __expf(Sf[tid][c] - mx);
        lsum += p;
        P[tid][c] = f2bf(p);
      }
      mrow[tid] = mx;
      lrow[tid] = lrow[tid] * corr + lsum;
      crow[tid] = corr;
    }
    __syncthreads();

    // ---- rescale O and accumulate O += P . V
#pragma unroll
    for (int r = 0; r < 8; ++r) {
      const float cf = crow[wm * 16 + r + hi * 8];
      o0[r] *= cf;
      o1[r] *= cf;
    }
#pragma unroll
    for (int kk = 0; kk < 2; ++kk) {
      v16bf pa = load_frag_a(&P[0][0], 72, wm * 16, kk * 32, lane);
      v16bf vb = load_frag_b(&Vt[0][0], 72, wn * 32, kk * 32, lane);
      o0 = wmma_bf16(pa, vb, o0);
      vb = load_frag_b(&Vt[0][0], 72, wn * 32 + 16, kk * 32, lane);
      o1 = wmma_bf16(pa, vb, o1);
    }
  }

  // ---- O /= l, write bf16 into [B*L, D] with head offset (no transpose needed)
#pragma unroll
  for (int r = 0; r < 8; ++r) {
    const int row = wm * 16 + r + hi * 8;
    const float inv = 1.0f / lrow[row];
    const size_t orow = (rowbase + q0 + row) * D_MODEL + h * HEAD_DIM;
    O[orow + wn * 32 + nn]      = f2bf(o0[r] * inv);
    O[orow + wn * 32 + 16 + nn] = f2bf(o1[r] * inv);
  }
}

// ---------------------------------------------------------------- launch
extern "C" void kernel_launch(void* const* d_in, const int* in_sizes, int n_in,
                              void* d_out, int out_size, void* d_ws,
                              size_t ws_size, hipStream_t stream) {
  (void)in_sizes; (void)n_in; (void)out_size; (void)ws_size;
  const float* x   = (const float*)d_in[0];
  const float* rpb = (const float*)d_in[1];
  const float* wq  = (const float*)d_in[2];
  const float* bq  = (const float*)d_in[3];
  const float* wk  = (const float*)d_in[4];
  const float* bk  = (const float*)d_in[5];
  const float* wv  = (const float*)d_in[6];
  const float* bv  = (const float*)d_in[7];
  const float* wo  = (const float*)d_in[8];
  const float* bo  = (const float*)d_in[9];
  const float* n1w = (const float*)d_in[10];
  const float* n2w = (const float*)d_in[11];
  const float* w1  = (const float*)d_in[12];
  const float* b1  = (const float*)d_in[13];
  const float* w2  = (const float*)d_in[14];
  const float* b2  = (const float*)d_in[15];
  float* out = (float*)d_out;

  // workspace layout (bytes)
  char* ws = (char*)d_ws;
  const size_t ACT = (size_t)NROWS * D_MODEL;   // 8M elements
  size_t off = 0;
  unsigned short* hbuf  = (unsigned short*)(ws + off); off += ACT * 2;
  unsigned short* qbuf  = (unsigned short*)(ws + off); off += ACT * 2;
  unsigned short* kbuf  = (unsigned short*)(ws + off); off += ACT * 2;
  unsigned short* vbuf  = (unsigned short*)(ws + off); off += ACT * 2;
  unsigned short* aobuf = (unsigned short*)(ws + off); off += ACT * 2;
  unsigned short* h2buf = (unsigned short*)(ws + off); off += ACT * 2;
  unsigned short* midbf = (unsigned short*)(ws + off); off += (size_t)NROWS * FFN_H * 2;
  float*          x2    = (float*)(ws + off);          off += ACT * 4;
  unsigned short* wqb   = (unsigned short*)(ws + off); off += (size_t)D_MODEL * D_MODEL * 2;
  unsigned short* wkb   = (unsigned short*)(ws + off); off += (size_t)D_MODEL * D_MODEL * 2;
  unsigned short* wvb   = (unsigned short*)(ws + off); off += (size_t)D_MODEL * D_MODEL * 2;
  unsigned short* wob   = (unsigned short*)(ws + off); off += (size_t)D_MODEL * D_MODEL * 2;
  unsigned short* w1b   = (unsigned short*)(ws + off); off += (size_t)FFN_H * D_MODEL * 2;
  unsigned short* w2b   = (unsigned short*)(ws + off); off += (size_t)D_MODEL * FFN_H * 2;

  const int DD = D_MODEL * D_MODEL;        // 1M
  const int DF = D_MODEL * FFN_H;          // 4M
  f32_to_bf16_kernel<<<DD / 256, 256, 0, stream>>>(wq, wqb, DD);
  f32_to_bf16_kernel<<<DD / 256, 256, 0, stream>>>(wk, wkb, DD);
  f32_to_bf16_kernel<<<DD / 256, 256, 0, stream>>>(wv, wvb, DD);
  f32_to_bf16_kernel<<<DD / 256, 256, 0, stream>>>(wo, wob, DD);
  f32_to_bf16_kernel<<<DF / 256, 256, 0, stream>>>(w1, w1b, DF);
  f32_to_bf16_kernel<<<DF / 256, 256, 0, stream>>>(w2, w2b, DF);

  // pre-norm 1
  rmsnorm_kernel<<<NROWS, 256, 0, stream>>>(x, n1w, hbuf);

  // QKV projections (bf16 out): grid = (M/128)*(N/64)
  const int qkv_blocks = (NROWS / 128) * (D_MODEL / 64);   // 1024
  gemm_bf16_tiled<1, 0, 0><<<qkv_blocks, 256, 0, stream>>>(
      hbuf, wqb, bq, nullptr, qbuf, NROWS, D_MODEL, D_MODEL);
  gemm_bf16_tiled<1, 0, 0><<<qkv_blocks, 256, 0, stream>>>(
      hbuf, wkb, bk, nullptr, kbuf, NROWS, D_MODEL, D_MODEL);
  gemm_bf16_tiled<1, 0, 0><<<qkv_blocks, 256, 0, stream>>>(
      hbuf, wvb, bv, nullptr, vbuf, NROWS, D_MODEL, D_MODEL);

  // flash attention
  dim3 ag(SEQL / 64, N_HEADS, BATCH);
  attn_kernel<<<ag, 256, 0, stream>>>(qbuf, kbuf, vbuf, rpb, aobuf);

  // out-proj + residual -> x2 (fp32)
  gemm_bf16_tiled<0, 0, 1><<<qkv_blocks, 256, 0, stream>>>(
      aobuf, wob, bo, x, x2, NROWS, D_MODEL, D_MODEL);

  // pre-norm 2
  rmsnorm_kernel<<<NROWS, 256, 0, stream>>>(x2, n2w, h2buf);

  // FFN: silu(h2 @ w1^T + b1) -> mid (bf16)
  const int f1_blocks = (NROWS / 128) * (FFN_H / 64);      // 4096
  gemm_bf16_tiled<1, 1, 0><<<f1_blocks, 256, 0, stream>>>(
      h2buf, w1b, b1, nullptr, midbf, NROWS, FFN_H, D_MODEL);

  // mid @ w2^T + b2 + x2 -> out (fp32)
  gemm_bf16_tiled<0, 0, 1><<<qkv_blocks, 256, 0, stream>>>(
      midbf, w2b, b2, x2, out, NROWS, D_MODEL, FFN_H);
}
